// RVMB_67688684585352
// MI455X (gfx1250) — matrix operand order repaired and checked by reading, not compiled, not run on hardware
//
#include <hip/hip_runtime.h>
#include <math.h>
#include <stdint.h>

#define LL   4096   // H*W
#define CC   96     // C
#define DINX 192    // DIN
#define KDIR 4      // K scan directions
#define NST  16     // N states
#define RRK  6      // R
#define CHX  38     // R + 2N
#define HHH  64
#define WWW  64

typedef __attribute__((ext_vector_type(16))) _Float16 v16h;
typedef __attribute__((ext_vector_type(8)))  float    v8f;

__device__ inline float wave_sum(float v) {
#pragma unroll
  for (int m = 16; m >= 1; m >>= 1) v += __shfl_xor(v, m, 32);
  return v;
}
__device__ inline float siluf(float x) { return x / (1.f + __expf(-x)); }
__device__ inline float softplusf(float x) { return x > 20.f ? x : log1pf(__expf(x)); }

// Load 8 contiguous f32 -> f16 fragment elements [base, base+8).
__device__ inline void load8h(const float* __restrict__ p, v16h& d, int base) {
  float4 x0 = *(const float4*)(p);
  float4 x1 = *(const float4*)(p + 4);
  d[base + 0] = (_Float16)x0.x; d[base + 1] = (_Float16)x0.y;
  d[base + 2] = (_Float16)x0.z; d[base + 3] = (_Float16)x0.w;
  d[base + 4] = (_Float16)x1.x; d[base + 5] = (_Float16)x1.y;
  d[base + 6] = (_Float16)x1.z; d[base + 7] = (_Float16)x1.w;
}

// ---------------------------------------------------------------------------
// GEMM: C[M,N] = A[M,K] * W[N,K]^T  (+ add1 + add2 when EPI), f16 WMMA / f32
// accumulate. One wave computes a 64(M) x 16(N) strip: W fragment is loaded
// once per K-chunk and feeds 4 WMMAs. M mult of 64, N mult of 16, K mult of 32.
// ---------------------------------------------------------------------------
template <bool EPI>
__global__ void gemm_abt_kernel(const float* __restrict__ A, const float* __restrict__ W,
                                float* __restrict__ C, int Kd, int N,
                                const float* __restrict__ add1,
                                const float* __restrict__ add2) {
  const int n0 = blockIdx.x * 16;
  const int m0 = blockIdx.y * 64;
  const int lane = threadIdx.x & 31;
  const int half = lane >> 4, lr = lane & 15;
  v8f acc[4] = {};
  for (int k0 = 0; k0 < Kd; k0 += 32) {
    v16h b;  // B[k][n] = W[n][k]; cols k0+half*16 .. +16 contiguous
    {
      const float* wr = W + (size_t)(n0 + lr) * Kd + k0 + half * 16;
      load8h(wr, b, 0);
      load8h(wr + 8, b, 8);
    }
#pragma unroll
    for (int s = 0; s < 4; ++s) {
      v16h a;  // A 16x32 layout: cols k0+half*16 .. +16 contiguous (matches B)
      const float* ar = A + (size_t)(m0 + s * 16 + lr) * Kd;
      load8h(ar + k0 + half * 16, a, 0);
      load8h(ar + k0 + half * 16 + 8, a, 8);
      acc[s] = __builtin_amdgcn_wmma_f32_16x16x32_f16(false, a, false, b, (short)0,
                                                      acc[s], false, false);
    }
  }
#pragma unroll
  for (int s = 0; s < 4; ++s)
#pragma unroll
    for (int r = 0; r < 8; ++r) {
      int m = m0 + s * 16 + r + half * 8, n = n0 + lr;
      float v = acc[s][r];
      if (EPI) v += add1[(size_t)m * N + n] + add2[(size_t)m * N + n];
      C[(size_t)m * N + n] = v;
    }
}

// ---------------------------------------------------------------------------
// GEMM: C[M,N] = A[M,K] * B[K,N] (both row-major), batched over blockIdx.z,
// with M-guard (used for x_proj: M=38).
// ---------------------------------------------------------------------------
__global__ void gemm_ab_kernel(const float* __restrict__ Abase,
                               const float* __restrict__ Bbase,
                               float* __restrict__ Cbase,
                               int M, int Kd, int N,
                               int strideA, int strideB, int strideC) {
  const float* A = Abase + (size_t)blockIdx.z * strideA;
  const float* B = Bbase + (size_t)blockIdx.z * strideB;
  float* C = Cbase + (size_t)blockIdx.z * strideC;
  const int m0 = blockIdx.y * 16, n0 = blockIdx.x * 16;
  const int lane = threadIdx.x & 31;
  const int half = lane >> 4, lr = lane & 15;
  const int row = m0 + lr;
  v8f acc = {};
  for (int k0 = 0; k0 < Kd; k0 += 32) {
    v16h a, b;
    if (row < M) {
      const float* ar = A + (size_t)row * Kd;
      load8h(ar + k0 + half * 16, a, 0);
      load8h(ar + k0 + half * 16 + 8, a, 8);
    } else {
#pragma unroll
      for (int e = 0; e < 16; ++e) a[e] = (_Float16)0.f;
    }
#pragma unroll
    for (int e = 0; e < 16; ++e)
      b[e] = (_Float16)B[(size_t)(k0 + e + half * 16) * N + (n0 + lr)];
    acc = __builtin_amdgcn_wmma_f32_16x16x32_f16(false, a, false, b, (short)0, acc,
                                                 false, false);
  }
#pragma unroll
  for (int r = 0; r < 8; ++r) {
    int m = m0 + r + half * 8, n = n0 + lr;
    if (m < M) C[(size_t)m * N + n] = acc[r];
  }
}

// ---------------------------------------------------------------------------
// Final 3x3 full conv as implicit GEMM: Out[l][co] = sum_{ci,ky,kx} X*Wr.
// K = 96*9 = 864 = 27*32. One wave = 64(M) x 16(N) strip; weight fragment
// gathered once per K-chunk, reused for 4 WMMAs. Epilogue: + bias + shortcut.
// ---------------------------------------------------------------------------
__global__ void conv_res_kernel(const float* __restrict__ X, const float* __restrict__ Wr,
                                const float* __restrict__ bias,
                                const float* __restrict__ sc, float* __restrict__ Out) {
  const int n0 = blockIdx.x * 16;
  const int m0 = blockIdx.y * 64;
  const int lane = threadIdx.x & 31;
  const int half = lane >> 4, lr = lane & 15;
  v8f acc[4] = {};
  for (int k0 = 0; k0 < 864; k0 += 32) {
    v16h b;
#pragma unroll
    for (int e = 0; e < 16; ++e) {
      int kk = k0 + e + half * 16;
      int ci = kk % CC, tap = kk / CC;
      b[e] = (_Float16)Wr[(size_t)(n0 + lr) * 864 + ci * 9 + tap];
    }
#pragma unroll
    for (int s = 0; s < 4; ++s) {
      v16h a;
#pragma unroll
      for (int e = 0; e < 16; ++e) {
        int kk = k0 + e + half * 16;
        int ci = kk % CC, tap = kk / CC;
        int ky = tap / 3, kx = tap % 3;
        int l = m0 + s * 16 + lr, h = l >> 6, w = l & 63;
        int hh = h + ky - 1, ww = w + kx - 1;
        float v = 0.f;
        if ((unsigned)hh < (unsigned)HHH && (unsigned)ww < (unsigned)WWW)
          v = X[(size_t)(hh * WWW + ww) * CC + ci];
        a[e] = (_Float16)v;
      }
      acc[s] = __builtin_amdgcn_wmma_f32_16x16x32_f16(false, a, false, b, (short)0,
                                                      acc[s], false, false);
    }
  }
#pragma unroll
  for (int s = 0; s < 4; ++s)
#pragma unroll
    for (int r = 0; r < 8; ++r) {
      int m = m0 + s * 16 + r + half * 8, n = n0 + lr;
      Out[(size_t)m * CC + n] = acc[s][r] + bias[n] + sc[(size_t)m * CC + n];
    }
}

// ---------------------------------------------------------------------------
// Double LayerNorm over C=96: xn = LN(x; g1,b1), xn2 = LN(xn; g2,b2).
// One wave per token row; lane handles 3 channels.
// ---------------------------------------------------------------------------
__global__ void ln2_kernel(const float* __restrict__ x,
                           const float* __restrict__ g1, const float* __restrict__ b1,
                           const float* __restrict__ g2, const float* __restrict__ b2,
                           float* __restrict__ xn, float* __restrict__ xn2) {
  int wid = (blockIdx.x * blockDim.x + threadIdx.x) >> 5;
  int lane = threadIdx.x & 31;
  if (wid >= LL) return;
  const float* xr = x + (size_t)wid * CC;
  float v[3], s = 0.f;
#pragma unroll
  for (int j = 0; j < 3; ++j) { v[j] = xr[lane * 3 + j]; s += v[j]; }
  s = wave_sum(s);
  float m = s * (1.f / CC), q = 0.f;
#pragma unroll
  for (int j = 0; j < 3; ++j) { float d = v[j] - m; q += d * d; }
  q = wave_sum(q);
  float inv = rsqrtf(q * (1.f / CC) + 1e-5f);
  float a[3]; s = 0.f;
#pragma unroll
  for (int j = 0; j < 3; ++j) {
    int c = lane * 3 + j;
    a[j] = (v[j] - m) * inv * g1[c] + b1[c];
    xn[(size_t)wid * CC + c] = a[j];
    s += a[j];
  }
  s = wave_sum(s); m = s * (1.f / CC); q = 0.f;
#pragma unroll
  for (int j = 0; j < 3; ++j) { float d = a[j] - m; q += d * d; }
  q = wave_sum(q);
  inv = rsqrtf(q * (1.f / CC) + 1e-5f);
#pragma unroll
  for (int j = 0; j < 3; ++j) {
    int c = lane * 3 + j;
    xn2[(size_t)wid * CC + c] = (a[j] - m) * inv * g2[c] + b2[c];
  }
}

// ---------------------------------------------------------------------------
// Depthwise 3x3 SAME conv + SiLU. xz is (L, 2*DIN) token-major; output x0 is
// channel-major (DIN, L).
// ---------------------------------------------------------------------------
__global__ void dwconv_kernel(const float* __restrict__ xz, const float* __restrict__ cw,
                              const float* __restrict__ cb, float* __restrict__ x0) {
  int idx = blockIdx.x * blockDim.x + threadIdx.x;
  if (idx >= DINX * LL) return;
  int l = idx % LL, d = idx / LL;
  int h = l >> 6, w = l & 63;
  float s = cb[d];
#pragma unroll
  for (int ky = 0; ky < 3; ++ky)
#pragma unroll
    for (int kx = 0; kx < 3; ++kx) {
      int hh = h + ky - 1, ww = w + kx - 1;
      if ((unsigned)hh < (unsigned)HHH && (unsigned)ww < (unsigned)WWW)
        s += xz[(size_t)(hh * WWW + ww) * (2 * DINX) + d] * cw[d * 9 + ky * 3 + kx];
    }
  x0[(size_t)d * LL + l] = siluf(s);
}

// ---------------------------------------------------------------------------
// Build the 4 scan directions xs[k][d][l]: identity, HW-transpose, flip, both.
// ---------------------------------------------------------------------------
__global__ void build_xs_kernel(const float* __restrict__ x0, float* __restrict__ xs) {
  int idx = blockIdx.x * blockDim.x + threadIdx.x;
  if (idx >= DINX * LL) return;
  int j = idx % LL, d = idx / LL;
  int p1 = (j & 63) * 64 + (j >> 6);
  int jr = LL - 1 - j;
  int p3 = (jr & 63) * 64 + (jr >> 6);
  const float* row = x0 + (size_t)d * LL;
  xs[((size_t)0 * DINX + d) * LL + j] = row[j];
  xs[((size_t)1 * DINX + d) * LL + j] = row[p1];
  xs[((size_t)2 * DINX + d) * LL + j] = row[jr];
  xs[((size_t)3 * DINX + d) * LL + j] = row[p3];
}

// ---------------------------------------------------------------------------
// delta = softplus(dt_w @ x_dbl[:R] + dt_b) (K dim = 6, plain VALU)
// ---------------------------------------------------------------------------
__global__ void dt_kernel(const float* __restrict__ xdbl, const float* __restrict__ dtw,
                          const float* __restrict__ dtb, float* __restrict__ delta) {
  int idx = blockIdx.x * blockDim.x + threadIdx.x;
  if (idx >= KDIR * DINX * LL) return;
  int l = idx % LL, kd = idx / LL, k = kd / DINX;
  float s = dtb[kd];
#pragma unroll
  for (int r = 0; r < RRK; ++r)
    s += dtw[kd * RRK + r] * xdbl[((size_t)k * CHX + r) * LL + l];
  delta[(size_t)kd * LL + l] = softplusf(s);
}

// ---------------------------------------------------------------------------
// Selective scan: 4 blocks (one per direction) x 192 threads (one per channel).
// h[16] in registers. Shared B/C staged in LDS via double-buffered
// GLOBAL_LOAD_ASYNC_TO_LDS_B128 (ASYNCcnt): next chunk's copy overlaps the
// current chunk's exp/fma chain. Per-thread streams prefetched.
// ---------------------------------------------------------------------------
#define STL 32
__global__ void scan_kernel(const float* __restrict__ delta, const float* __restrict__ xs,
                            const float* __restrict__ alog, const float* __restrict__ xdbl,
                            const float* __restrict__ Ds, float* __restrict__ ys) {
  int k = blockIdx.x;
  int d = threadIdx.x;
  int kd = k * DINX + d;
  float Ar[NST];
#pragma unroll
  for (int n = 0; n < NST; ++n) Ar[n] = -__expf(alog[(size_t)kd * NST + n]);
  float h[NST];
#pragma unroll
  for (int n = 0; n < NST; ++n) h[n] = 0.f;
  const float* Bp = xdbl + ((size_t)k * CHX + RRK) * LL;
  const float* Cp = xdbl + ((size_t)k * CHX + RRK + NST) * LL;
  const float* dp = delta + (size_t)kd * LL;
  const float* up = xs + (size_t)kd * LL;
  float* yp = ys + (size_t)kd * LL;
  float Dv = Ds[kd];

  // [buf][B/C][n][t]
  __shared__ float sBC[2][2][NST][STL];

  // Issue async global->LDS copies for one 32-step chunk (2 x 16 x 32 floats =
  // 4KB = 256 x 16B lane-transfers, striped over 192 threads).
  auto issue_chunk = [&](int l0, int buf) {
    for (int s = threadIdx.x; s < 256; s += DINX) {
      int isC = s >> 7;         // 0 = B rows, 1 = C rows
      int slot = s & 127;       // 16 rows x 8 x 16B segments
      int n = slot >> 3, seg = slot & 7;
      const float* gp = (isC ? Cp : Bp) + (size_t)n * LL + l0 + seg * 4;
      uint32_t lds = (uint32_t)(uintptr_t)&sBC[buf][isC][n][seg * 4];
      uint64_t ga = (uint64_t)(uintptr_t)gp;
      asm volatile("global_load_async_to_lds_b128 %0, %1, off"
                   :: "v"(lds), "v"(ga) : "memory");
    }
  };

  issue_chunk(0, 0);
  asm volatile("s_wait_asynccnt 0x0" ::: "memory");
  __syncthreads();

  for (int l0 = 0; l0 < LL; l0 += STL) {
    int buf = (l0 / STL) & 1;
    if (l0 + STL < LL) {
      issue_chunk(l0 + STL, buf ^ 1);
      __builtin_prefetch(dp + l0 + STL, 0, 0);
      __builtin_prefetch(up + l0 + STL, 0, 0);
    }
    const float(*sB)[STL] = sBC[buf][0];
    const float(*sC)[STL] = sBC[buf][1];
    for (int t = 0; t < STL; ++t) {
      int l = l0 + t;
      float dl = dp[l], uv = up[l];
      float du = dl * uv, y = 0.f;
#pragma unroll
      for (int n = 0; n < NST; ++n) {
        h[n] = __expf(dl * Ar[n]) * h[n] + du * sB[n][t];
        y += h[n] * sC[n][t];
      }
      yp[l] = y + Dv * uv;
    }
    asm volatile("s_wait_asynccnt 0x0" ::: "memory");
    __syncthreads();
  }
}

// ---------------------------------------------------------------------------
// Combine 4 directions (undo flips/transposes), LayerNorm over DIN, *silu(z).
// One wave per token row; lane handles 6 channels.
// ---------------------------------------------------------------------------
__global__ void combine_kernel(const float* __restrict__ ys, const float* __restrict__ xz,
                               const float* __restrict__ og, const float* __restrict__ ob,
                               float* __restrict__ yg) {
  int wid = (blockIdx.x * blockDim.x + threadIdx.x) >> 5;
  int lane = threadIdx.x & 31;
  if (wid >= LL) return;
  int l = wid;
  int j1 = (l & 63) * 64 + (l >> 6);
  int l2 = LL - 1 - l;
  int j3 = LL - 1 - j1;
  float v[6], s = 0.f;
#pragma unroll
  for (int jj = 0; jj < 6; ++jj) {
    int d = lane * 6 + jj;
    float t = ys[((size_t)0 * DINX + d) * LL + l] + ys[((size_t)1 * DINX + d) * LL + j1] +
              ys[((size_t)2 * DINX + d) * LL + l2] + ys[((size_t)3 * DINX + d) * LL + j3];
    v[jj] = t; s += t;
  }
  s = wave_sum(s);
  float m = s * (1.f / DINX), q = 0.f;
#pragma unroll
  for (int jj = 0; jj < 6; ++jj) { float dd = v[jj] - m; q += dd * dd; }
  q = wave_sum(q);
  float inv = rsqrtf(q * (1.f / DINX) + 1e-5f);
#pragma unroll
  for (int jj = 0; jj < 6; ++jj) {
    int d = lane * 6 + jj;
    float z = xz[(size_t)l * (2 * DINX) + DINX + d];
    yg[(size_t)l * DINX + d] = ((v[jj] - m) * inv * og[d] + ob[d]) * siluf(z);
  }
}

extern "C" void kernel_launch(void* const* d_in, const int* in_sizes, int n_in,
                              void* d_out, int out_size, void* d_ws, size_t ws_size,
                              hipStream_t stream) {
  (void)in_sizes; (void)n_in; (void)out_size; (void)ws_size;
  const float* x    = (const float*)d_in[0];
  const float* n1g  = (const float*)d_in[3];
  const float* n1b  = (const float*)d_in[4];
  const float* vng  = (const float*)d_in[5];
  const float* vnb  = (const float*)d_in[6];
  const float* ipw  = (const float*)d_in[7];
  const float* cw   = (const float*)d_in[8];
  const float* cb   = (const float*)d_in[9];
  const float* xpw  = (const float*)d_in[10];
  const float* dtw  = (const float*)d_in[11];
  const float* dtb  = (const float*)d_in[12];
  const float* alog = (const float*)d_in[13];
  const float* dsp  = (const float*)d_in[14];
  const float* ong  = (const float*)d_in[15];
  const float* onb  = (const float*)d_in[16];
  const float* opw  = (const float*)d_in[17];
  const float* rcw  = (const float*)d_in[18];
  const float* rcb  = (const float*)d_in[19];

  float* ws = (float*)d_ws;
  size_t off = 0;
  auto carve = [&](size_t n) { float* p = ws + off; off += n; return p; };
  float* xn    = carve((size_t)LL * CC);
  float* xn2   = carve((size_t)LL * CC);
  float* xz    = carve((size_t)LL * 2 * DINX);
  float* x0c   = carve((size_t)DINX * LL);
  float* xsb   = carve((size_t)KDIR * DINX * LL);
  float* xdbl  = carve((size_t)KDIR * CHX * LL);
  float* delta = carve((size_t)KDIR * DINX * LL);
  float* ysb   = carve((size_t)KDIR * DINX * LL);
  float* yg    = carve((size_t)LL * DINX);
  float* xb0   = carve((size_t)LL * CC);
  float* xb1   = carve((size_t)LL * CC);

  const float* xcur = x;
  float* xout[2] = {xb0, xb1};
  for (int i = 0; i < 2; ++i) {
    ln2_kernel<<<512, 256, 0, stream>>>(xcur, n1g + i * CC, n1b + i * CC,
                                        vng + i * CC, vnb + i * CC, xn, xn2);
    // in_proj: (4096x96) @ (384x96)^T -> xz (4096x384)
    gemm_abt_kernel<false><<<dim3(24, 64), 32, 0, stream>>>(
        xn2, ipw + (size_t)i * 2 * DINX * CC, xz, CC, 2 * DINX, nullptr, nullptr);
    dwconv_kernel<<<(DINX * LL) / 256, 256, 0, stream>>>(
        xz, cw + (size_t)i * DINX * 9, cb + (size_t)i * DINX, x0c);
    build_xs_kernel<<<(DINX * LL) / 256, 256, 0, stream>>>(x0c, xsb);
    // x_proj per direction: (38x192) @ (192x4096) -> xdbl[k]
    gemm_ab_kernel<<<dim3(256, 3, KDIR), 32, 0, stream>>>(
        xpw + (size_t)i * KDIR * CHX * DINX, xsb, xdbl, CHX, DINX, LL,
        CHX * DINX, DINX * LL, CHX * LL);
    dt_kernel<<<(KDIR * DINX * LL) / 256, 256, 0, stream>>>(
        xdbl, dtw + (size_t)i * KDIR * DINX * RRK, dtb + (size_t)i * KDIR * DINX, delta);
    scan_kernel<<<KDIR, DINX, 0, stream>>>(
        delta, xsb, alog + (size_t)i * KDIR * DINX * NST, xdbl,
        dsp + (size_t)i * KDIR * DINX, ysb);
    combine_kernel<<<512, 256, 0, stream>>>(ysb, xz, ong + i * DINX, onb + i * DINX, yg);
    // out_proj: (4096x192) @ (96x192)^T -> next x, fused + xn + x_prev
    gemm_abt_kernel<true><<<dim3(6, 64), 32, 0, stream>>>(
        yg, opw + (size_t)i * CC * DINX, xout[i], DINX, CC, xn, xcur);
    xcur = xout[i];
  }
  conv_res_kernel<<<dim3(6, 64), 32, 0, stream>>>(xcur, rcw, rcb, x, (float*)d_out);
}
